// NasAutoGraphB_36816459661707
// MI455X (gfx1250) — compile-verified, matrix operand order checked by reading, not run on hardware
//
#include <hip/hip_runtime.h>
#include <math.h>

// ---------------------------------------------------------------------------
// Types for CDNA5 WMMA (wave32): bf16 A/B operands, f32 accumulator.
// ---------------------------------------------------------------------------
typedef __attribute__((ext_vector_type(16))) __bf16          v16bf;
typedef __attribute__((ext_vector_type(16))) unsigned short  v16u;
typedef __attribute__((ext_vector_type(8)))  float           v8f;

union BfVec { v16u u; v16bf b; };

__device__ __forceinline__ unsigned short f2bf_bits(float f) {
  union { __bf16 b; unsigned short u; } cv;
  cv.b = (__bf16)f;                      // hardware f32->bf16 convert
  return cv.u;
}

// ---------------------------------------------------------------------------
// Utility kernels
// ---------------------------------------------------------------------------
__global__ void zero_kernel(float* __restrict__ p, long n) {
  long i = (long)blockIdx.x * blockDim.x + threadIdx.x;
  long stride = (long)gridDim.x * blockDim.x;
  for (; i < n; i += stride) p[i] = 0.0f;
}

__global__ void deg_cnt_kernel(const int* __restrict__ col, const float* __restrict__ ew,
                               float* __restrict__ deg, float* __restrict__ cnt, long E) {
  long e = (long)blockIdx.x * blockDim.x + threadIdx.x;
  if (e >= E) return;
  int c = col[e];
  atomicAdd(&deg[c], ew[e]);
  atomicAdd(&cnt[c], 1.0f);
}

// deg -> dinv (in place), cnt -> 1/max(cnt,1) (in place)
__global__ void dinv_cntinv_kernel(float* __restrict__ deg, float* __restrict__ cnt, long n) {
  long i = (long)blockIdx.x * blockDim.x + threadIdx.x;
  if (i >= n) return;
  float d = deg[i];
  deg[i] = (d > 0.0f) ? rsqrtf(d) : 0.0f;
  float c = cnt[i];
  cnt[i] = 1.0f / fmaxf(c, 1.0f);
}

__global__ void norm_kernel(const int* __restrict__ row, const int* __restrict__ col,
                            const float* __restrict__ ew, const float* __restrict__ dinv,
                            float* __restrict__ nrm, long E) {
  long e = (long)blockIdx.x * blockDim.x + threadIdx.x;
  if (e >= E) return;
  nrm[e] = dinv[row[e]] * ew[e] * dinv[col[e]];
}

// ---------------------------------------------------------------------------
// WMMA GEMM:  Out[M x (NCT*16)] = A[M x Kin] (row-scaled) @ W[Kin x Kout] + bias
//
// 256-thread block = 8 waves; each wave computes a 16-row strip across all NCT
// column tiles. Weights are converted to bf16 AND pre-packed into per-lane
// WMMA B-fragment order in LDS, so the inner loop loads each fragment as one
// contiguous 32-byte LDS read (2x ds_load_b128). All NCT fragments for a
// k-step are loaded before the WMMA burst so a single dscnt wait covers them.
//
// Fragment layout in sW (u16 units):
//   frag = (k0/32)*NCT + ct ; offset = frag*512 + lane*16 + e
//   element (lane, e) = W[kk][ncol],
//     kk   = k0 + ((lane>=16)?8:0) + (e<8 ? e : e+8)
//     ncol = ct*16 + (lane&15)
// ---------------------------------------------------------------------------
template <int NCT>
__global__ void gemm_wmma_kernel(const float* __restrict__ A, int lda,
                                 const float* __restrict__ W,       // Kin x Kout, row-major f32
                                 const float* __restrict__ bias,    // Kout or null
                                 const float* __restrict__ rscale,  // per-row A scale or null
                                 float* __restrict__ Out, int ldo,
                                 int M, int Kin, int accum)
{
  constexpr int Kout = NCT * 16;
  __shared__ alignas(32) unsigned short sW[NCT * 4096];  // (256/32)*NCT*512, <=64KB

  const int tid = threadIdx.x;
  // Fill LDS with W in bf16 fragment order (gather happens once per block).
  const int nFill = (Kin >> 5) * NCT * 512;
  for (int idx = tid; idx < nFill; idx += 256) {
    const int e    = idx & 15;
    const int l    = (idx >> 4) & 31;
    const int frag = idx >> 9;
    const int ct   = frag % NCT;
    const int ks   = frag / NCT;
    const int kk   = (ks << 5) + ((l >= 16) ? 8 : 0) + (e < 8 ? e : e + 8);
    const int ncol = ct * 16 + (l & 15);
    sW[idx] = f2bf_bits(W[kk * Kout + ncol]);
  }
  __syncthreads();

  const int  wave    = tid >> 5;
  const int  lane    = tid & 31;
  const long rowBase = ((long)blockIdx.x * 8 + wave) * 16;
  if (rowBase >= M) return;

  const int  m      = lane & 15;
  const long arow   = rowBase + m;
  const bool rvalid = (arow < (long)M);
  const float* Arow = A + (rvalid ? arow * (long)lda : 0);
  // rs==0 for out-of-range rows: loads stay in-bounds (row 0) but contribute 0.
  float rs = rvalid ? ((rscale != nullptr) ? rscale[arow] : 1.0f) : 0.0f;
  const int khalf = (lane >= 16) ? 8 : 0;

  v8f acc[NCT];
#pragma unroll
  for (int ct = 0; ct < NCT; ++ct)
#pragma unroll
    for (int r = 0; r < 8; ++r) acc[ct][r] = 0.0f;

  for (int k0 = 0; k0 < Kin; k0 += 32) {
    // A fragment: two contiguous 8-float K segments, 32B-aligned.
    const float4 a0 = *(const float4*)(Arow + k0 + khalf + 0);
    const float4 a1 = *(const float4*)(Arow + k0 + khalf + 4);
    const float4 a2 = *(const float4*)(Arow + k0 + khalf + 16);
    const float4 a3 = *(const float4*)(Arow + k0 + khalf + 20);
    BfVec a;
    a.b[0]  = (__bf16)(a0.x * rs); a.b[1]  = (__bf16)(a0.y * rs);
    a.b[2]  = (__bf16)(a0.z * rs); a.b[3]  = (__bf16)(a0.w * rs);
    a.b[4]  = (__bf16)(a1.x * rs); a.b[5]  = (__bf16)(a1.y * rs);
    a.b[6]  = (__bf16)(a1.z * rs); a.b[7]  = (__bf16)(a1.w * rs);
    a.b[8]  = (__bf16)(a2.x * rs); a.b[9]  = (__bf16)(a2.y * rs);
    a.b[10] = (__bf16)(a2.z * rs); a.b[11] = (__bf16)(a2.w * rs);
    a.b[12] = (__bf16)(a3.x * rs); a.b[13] = (__bf16)(a3.y * rs);
    a.b[14] = (__bf16)(a3.z * rs); a.b[15] = (__bf16)(a3.w * rs);

    const unsigned short* fbase = sW + (((k0 >> 5) * NCT) << 9) + (lane << 4);
    // Load all B fragments for this k-step first (one dscnt wait for all),
    // then issue the WMMA burst.
    BfVec bfr[NCT];
#pragma unroll
    for (int ct = 0; ct < NCT; ++ct)
      bfr[ct].u = *(const v16u*)(fbase + (ct << 9));   // 32B -> 2x ds_load_b128
#pragma unroll
    for (int ct = 0; ct < NCT; ++ct)
      acc[ct] = __builtin_amdgcn_wmma_f32_16x16x32_bf16(
          false, a.b, false, bfr[ct].b, (short)0, acc[ct], false, false);
  }

  // C/D layout: lane holds N=lane%16; VGPR r holds M = r + 8*(lane>=16)
  const int mhi = (lane >> 4) << 3;
#pragma unroll
  for (int ct = 0; ct < NCT; ++ct) {
    const int ncol = ct * 16 + (lane & 15);
    const float bv = (bias != nullptr) ? bias[ncol] : 0.0f;
#pragma unroll
    for (int r = 0; r < 8; ++r) {
      const long orow = rowBase + mhi + r;
      if (orow < (long)M) {
        float v = acc[ct][r] + bv;
        float* p = Out + orow * (long)ldo + ncol;
        if (accum) v += *p;
        *p = v;
      }
    }
  }
}

// ---------------------------------------------------------------------------
// Fused edge kernel: 32 lanes per edge, float4 per lane.
//   agg[col] += norm * t[row]   (ARMA propagation)
//   s[col]   += ew   * h[row]   (SAGE weighted sum)
// ---------------------------------------------------------------------------
__global__ void edge_scatter_kernel(const int* __restrict__ row, const int* __restrict__ col,
                                    const float* __restrict__ nrm, const float* __restrict__ ew,
                                    const float* __restrict__ t, const float* __restrict__ h,
                                    float* __restrict__ agg, float* __restrict__ s, long E)
{
  long idx = (long)blockIdx.x * blockDim.x + threadIdx.x;
  if (idx >= E * 32) return;
  long e  = idx >> 5;
  int  f0 = (int)(idx & 31) * 4;
  int  r  = row[e], c = col[e];
  float nm = nrm[e], w = ew[e];
  const float4 tv = *(const float4*)(t + (long)r * 128 + f0);
  const float4 hv = *(const float4*)(h + (long)r * 128 + f0);
  float* ap = agg + (long)c * 128 + f0;
  float* sp = s   + (long)c * 128 + f0;
  atomicAdd(ap + 0, nm * tv.x); atomicAdd(ap + 1, nm * tv.y);
  atomicAdd(ap + 2, nm * tv.z); atomicAdd(ap + 3, nm * tv.w);
  atomicAdd(sp + 0, w * hv.x);  atomicAdd(sp + 1, w * hv.y);
  atomicAdd(sp + 2, w * hv.z);  atomicAdd(sp + 3, w * hv.w);
}

// hcat[:, 0:128] = elu(leaky(relu(agg + hv))) = relu(agg + hv)
__global__ void arma_post_kernel(const float* __restrict__ agg, const float* __restrict__ hv,
                                 float* __restrict__ hcat, long n128) {
  long i = (long)blockIdx.x * blockDim.x + threadIdx.x;
  if (i >= n128) return;
  float v = fmaxf(agg[i] + hv[i], 0.0f);
  hcat[(i >> 7) * 256 + (i & 127)] = v;
}

// hcat[:, 128:256] = elu(leaky_relu(so, 0.01))
__global__ void sage_post_kernel(const float* __restrict__ so,
                                 float* __restrict__ hcat, long n128) {
  long i = (long)blockIdx.x * blockDim.x + threadIdx.x;
  if (i >= n128) return;
  float v = so[i];
  v = (v >= 0.0f) ? v : 0.01f * v;            // leaky relu
  v = (v >= 0.0f) ? v : (expf(v) - 1.0f);     // elu
  hcat[(i >> 7) * 256 + 128 + (i & 127)] = v;
}

__global__ void logsoftmax_kernel(const float* __restrict__ logits,
                                  float* __restrict__ out, long Nn) {
  long i = (long)blockIdx.x * blockDim.x + threadIdx.x;
  if (i >= Nn) return;
  const float* p = logits + i * 16;
  float m = p[0];
#pragma unroll
  for (int j = 1; j < 16; ++j) m = fmaxf(m, p[j]);
  float sum = 0.0f;
#pragma unroll
  for (int j = 0; j < 16; ++j) sum += expf(p[j] - m);
  float l = logf(sum);
  float* o = out + i * 16;
#pragma unroll
  for (int j = 0; j < 16; ++j) o[j] = p[j] - m - l;
}

// ---------------------------------------------------------------------------
// Launch
// ---------------------------------------------------------------------------
extern "C" void kernel_launch(void* const* d_in, const int* in_sizes, int n_in,
                              void* d_out, int out_size, void* d_ws, size_t ws_size,
                              hipStream_t stream)
{
  (void)n_in; (void)out_size; (void)ws_size;
  const float* x        = (const float*)d_in[0];
  const int*   ei       = (const int*)  d_in[1];
  const float* ew       = (const float*)d_in[2];
  const float* pre_w0   = (const float*)d_in[3];
  const float* pre_b0   = (const float*)d_in[4];
  const float* arma_w0  = (const float*)d_in[5];
  const float* arma_v0  = (const float*)d_in[6];
  const float* arma_b0  = (const float*)d_in[7];
  const float* sage_wl0 = (const float*)d_in[8];
  const float* sage_bl0 = (const float*)d_in[9];
  const float* sage_wr0 = (const float*)d_in[10];
  const float* pre_w1   = (const float*)d_in[11];
  const float* pre_b1   = (const float*)d_in[12];
  const float* arma_w1  = (const float*)d_in[13];
  const float* arma_v1  = (const float*)d_in[14];
  const float* arma_b1  = (const float*)d_in[15];
  const float* sage_wl1 = (const float*)d_in[16];
  const float* sage_bl1 = (const float*)d_in[17];
  const float* sage_wr1 = (const float*)d_in[18];
  const float* cls_w    = (const float*)d_in[19];
  const float* cls_b    = (const float*)d_in[20];

  const long N = in_sizes[0] / 128;
  const long E = in_sizes[2];
  const int* row = ei;       // edge_index[0]
  const int* col = ei + E;   // edge_index[1]

  // Workspace carve (floats)
  float* ws = (float*)d_ws;
  float* deg   = ws; ws += N;        // -> dinv in place
  float* cnt   = ws; ws += N;        // -> 1/max(cnt,1) in place
  float* nrm   = ws; ws += E;
  float* hpre  = ws; ws += N * 128;
  float* tbuf  = ws; ws += N * 128;  // t, then reused as hv
  float* agg   = ws; ws += N * 128;  // agg, then reused as SAGE pre-activation
  float* sbuf  = ws; ws += N * 128;  // must stay adjacent to agg (zeroed together)
  float* hcat0 = ws; ws += N * 256;
  float* hcat1 = ws; ws += N * 256;
  float* logits = ws;

  auto cdiv = [](long a, long b) { return (int)((a + b - 1) / b); };
  const dim3 blk(256);
  const int gemmBlocks = cdiv(N, 128);

  // --- graph normalization ---
  zero_kernel<<<cdiv(2 * N, 256), blk, 0, stream>>>(deg, 2 * N);   // deg + cnt
  deg_cnt_kernel<<<cdiv(E, 256), blk, 0, stream>>>(col, ew, deg, cnt, E);
  dinv_cntinv_kernel<<<cdiv(N, 256), blk, 0, stream>>>(deg, cnt, N);
  norm_kernel<<<cdiv(E, 256), blk, 0, stream>>>(row, col, ew, deg, nrm, E);

  const float* cellW[2][8] = {
    {pre_w0, pre_b0, arma_w0, arma_v0, arma_b0, sage_wl0, sage_bl0, sage_wr0},
    {pre_w1, pre_b1, arma_w1, arma_v1, arma_b1, sage_wl1, sage_bl1, sage_wr1},
  };
  const float* hin  = x;     int ldin = 128;  int kin = 128;
  float*       hout = hcat0;

  for (int cell = 0; cell < 2; ++cell) {
    const float* pw  = cellW[cell][0]; const float* pb  = cellW[cell][1];
    const float* aw  = cellW[cell][2]; const float* av  = cellW[cell][3];
    const float* ab  = cellW[cell][4];
    const float* swl = cellW[cell][5]; const float* sbl = cellW[cell][6];
    const float* swr = cellW[cell][7];

    // h_pre = hin @ pw + pb
    gemm_wmma_kernel<8><<<gemmBlocks, blk, 0, stream>>>(hin, ldin, pw, pb, nullptr,
                                                        hpre, 128, (int)N, kin, 0);
    // t = h_pre @ arma_w
    gemm_wmma_kernel<8><<<gemmBlocks, blk, 0, stream>>>(hpre, 128, aw, nullptr, nullptr,
                                                        tbuf, 128, (int)N, 128, 0);
    // agg = 0, s = 0 ; fused edge scatter
    zero_kernel<<<cdiv(2 * N * 128, 256), blk, 0, stream>>>(agg, 2 * N * 128);
    edge_scatter_kernel<<<cdiv(E * 32, 256), blk, 0, stream>>>(row, col, nrm, ew,
                                                               tbuf, hpre, agg, sbuf, E);
    // hv = h_pre @ arma_v + b  (reuse tbuf) ; hcat[:, :128] = relu(agg + hv)
    gemm_wmma_kernel<8><<<gemmBlocks, blk, 0, stream>>>(hpre, 128, av, ab, nullptr,
                                                        tbuf, 128, (int)N, 128, 0);
    arma_post_kernel<<<cdiv(N * 128, 256), blk, 0, stream>>>(agg, tbuf, hout, N * 128);
    // so = (s/cnt) @ wl + bl ; so += h_pre @ wr ; hcat[:, 128:] = elu(leaky(so))
    gemm_wmma_kernel<8><<<gemmBlocks, blk, 0, stream>>>(sbuf, 128, swl, sbl, cnt,
                                                        agg, 128, (int)N, 128, 0);
    gemm_wmma_kernel<8><<<gemmBlocks, blk, 0, stream>>>(hpre, 128, swr, nullptr, nullptr,
                                                        agg, 128, (int)N, 128, 1);
    sage_post_kernel<<<cdiv(N * 128, 256), blk, 0, stream>>>(agg, hout, N * 128);

    hin = hout; ldin = 256; kin = 256;
    hout = hcat1;
  }

  // classifier (Kout = 16 -> NCT = 1) + log_softmax
  gemm_wmma_kernel<1><<<gemmBlocks, blk, 0, stream>>>(hcat1, 256, cls_w, cls_b, nullptr,
                                                      logits, 16, (int)N, 256, 0);
  logsoftmax_kernel<<<cdiv(N, 256), blk, 0, stream>>>(logits, (float*)d_out, N);
}
